// AttentionBlock_32925219291214
// MI455X (gfx1250) — matrix operand order repaired and checked by reading, not compiled, not run on hardware
//
#include <hip/hip_runtime.h>
#include <cstdint>

typedef __attribute__((ext_vector_type(16))) _Float16 v16h;
typedef __attribute__((ext_vector_type(8)))  _Float16 v8h;
typedef __attribute__((ext_vector_type(4)))  _Float16 v4h;
typedef __attribute__((ext_vector_type(8)))  float    v8f;
typedef __attribute__((ext_vector_type(4)))  int      v4i;

#define NB_   8
#define NC_   512
#define NT_   1024
#define NHEAD 8
#define NCH   64
#define NO3   1536   // 3*C

// ---------------- async global -> LDS copy (gfx1250 ASYNCcnt path) ----------------
#if defined(__has_builtin)
#if __has_builtin(__builtin_amdgcn_global_load_async_to_lds_b128) && \
    __has_builtin(__builtin_amdgcn_s_wait_asynccnt)
#define HAS_ASYNC 1
#endif
#endif
#ifndef HAS_ASYNC
#define HAS_ASYNC 0
#endif

#if HAS_ASYNC
typedef __attribute__((address_space(1))) v4i as1_v4i;   // global int4
typedef __attribute__((address_space(3))) v4i as3_v4i;   // LDS int4
#define WAIT_ASYNC(n) __builtin_amdgcn_s_wait_asynccnt(n)
#else
#define WAIT_ASYNC(n) ((void)0)
#endif

// copy 16 bytes global -> LDS (async when available)
__device__ __forceinline__ void g2l_b128(_Float16* l, const _Float16* g) {
#if HAS_ASYNC
  __builtin_amdgcn_global_load_async_to_lds_b128(
      (as1_v4i*)(unsigned long long)(const void*)g,
      (as3_v4i*)(unsigned long long)(void*)l, 0, 0);
#else
  *(v8h*)l = *(const v8h*)g;
#endif
}

// ---------------- WMMA helpers (gfx1250 wave32) ----------------

__device__ __forceinline__ v8f wmma_f16(v16h a, v16h b, v8f c) {
  // D = A(16x32 f16) * B(32x16 f16) + C(16x16 f32)
  return __builtin_amdgcn_wmma_f32_16x16x32_f16(false, a, false, b, (short)0, c, false, false);
}

// A fragment: row = lane&15; lanes 0-15 hold k = kb+[0..7],[16..23],
// lanes 16-31 hold k = kb+[8..15],[24..31].
__device__ __forceinline__ v16h ldA(const _Float16* row, int kb, int hi) {
  v8h a0 = *(const v8h*)(row + kb + hi * 8);
  v8h a1 = *(const v8h*)(row + kb + hi * 8 + 16);
  v16h r;
#pragma unroll
  for (int i = 0; i < 8; ++i) { r[i] = a0[i]; r[i + 8] = a1[i]; }
  return r;
}

// B fragment ([n][k] in LDS): col n = lane&15; lanes 0-15: k = kb+[0..7],[16..23],
// lanes 16-31: k = kb+[8..15],[24..31].
__device__ __forceinline__ v16h ldB(const _Float16* row, int kb, int hi) {
  v8h b0 = *(const v8h*)(row + kb + hi * 8);
  v8h b1 = *(const v8h*)(row + kb + hi * 8 + 16);
  v16h r;
#pragma unroll
  for (int i = 0; i < 8; ++i) { r[i] = b0[i]; r[i + 8] = b1[i]; }
  return r;
}

// ---------------- Kernel 1: GroupNorm -> xnT[b][t][c] (f16) ----------------
__global__ void gn_kernel(const float* __restrict__ x, const float* __restrict__ w,
                          const float* __restrict__ bias, _Float16* __restrict__ xnT) {
  int b = blockIdx.x >> 5;
  int g = blockIdx.x & 31;
  const float* xb = x + ((size_t)(b * NC_ + g * 16)) * NT_;
  float s = 0.f, s2 = 0.f;
  for (int i = threadIdx.x; i < 16 * NT_; i += 256) {
    float v = xb[i];
    s += v; s2 += v * v;
  }
  __shared__ float rs[256], rq[256];
  rs[threadIdx.x] = s; rq[threadIdx.x] = s2;
  __syncthreads();
  for (int off = 128; off > 0; off >>= 1) {
    if (threadIdx.x < off) {
      rs[threadIdx.x] += rs[threadIdx.x + off];
      rq[threadIdx.x] += rq[threadIdx.x + off];
    }
    __syncthreads();
  }
  const float inv_n = 1.0f / 16384.0f;
  float mu  = rs[0] * inv_n;
  float var = rq[0] * inv_n - mu * mu;
  float rsig = rsqrtf(var + 1e-5f);
  for (int i = threadIdx.x; i < 16 * NT_; i += 256) {
    int cl = i >> 10;
    int t  = i & (NT_ - 1);
    int c  = g * 16 + cl;
    float v = (xb[i] - mu) * rsig * w[c] + bias[c];
    xnT[((size_t)(b * NT_ + t)) * NC_ + c] = (_Float16)v;
  }
}

// ---------------- Kernel 2: f32 -> f16 convert ----------------
__global__ void cvt_kernel(const float* __restrict__ in, _Float16* __restrict__ out, int n) {
  int i = blockIdx.x * 256 + threadIdx.x;
  if (i < n) out[i] = (_Float16)in[i];
}

// ---------------- Kernel 3: qkv GEMM (double-buffered, async tiles) ----------------
// qkvT[b][t][o] = sum_c xnT[b][t][c] * qw[o][c] + qb[o]
__global__ void qkv_gemm(const _Float16* __restrict__ xnT, const _Float16* __restrict__ qw,
                         const float* __restrict__ qb, _Float16* __restrict__ qkvT) {
  __shared__ __align__(16) _Float16 As[2][64][64];   // [m][k]
  __shared__ __align__(16) _Float16 Bs[2][64][64];   // [n][k]
  int n0 = blockIdx.x * 64, m0 = blockIdx.y * 64, b = blockIdx.z;
  int tid = threadIdx.x, wave = tid >> 5, lane = tid & 31, l = lane & 15, hi = lane >> 4;
  int mb = (wave >> 1) * 16, nb0 = (wave & 1) * 32;
  const _Float16* Ag = xnT + ((size_t)(b * NT_ + m0)) * NC_;
  const _Float16* Bg = qw + (size_t)n0 * NC_;

  // 512 b128-chunks per matrix; 256 threads -> 2 chunks each per matrix (4 async loads)
  auto issue = [&](int buf, int k0) {
#pragma unroll
    for (int idx = tid; idx < 512; idx += 256) {
      int r = idx >> 3, c8 = (idx & 7) * 8;
      g2l_b128(&As[buf][r][c8], Ag + (size_t)r * NC_ + k0 + c8);
      g2l_b128(&Bs[buf][r][c8], Bg + (size_t)r * NC_ + k0 + c8);
    }
  };

  issue(0, 0);
  v8f c0 = {}, c1 = {};
  const int NK = NC_ / 64;                     // 8 k-tiles
  for (int kt = 0; kt < NK; ++kt) {
    int cur = kt & 1;
    if (kt + 1 < NK) { issue(cur ^ 1, (kt + 1) * 64); WAIT_ASYNC(4); }
    else             { WAIT_ASYNC(0); }
    __syncthreads();                           // tile kt visible to all waves
    const _Float16* arow  = &As[cur][mb + l][0];
    const _Float16* brow0 = &Bs[cur][nb0 + l][0];
    const _Float16* brow1 = &Bs[cur][nb0 + 16 + l][0];
#pragma unroll
    for (int kk = 0; kk < 64; kk += 32) {
      v16h af  = ldA(arow, kk, hi);
      v16h bf0 = ldB(brow0, kk, hi);
      v16h bf1 = ldB(brow1, kk, hi);
      c0 = wmma_f16(af, bf0, c0);
      c1 = wmma_f16(af, bf1, c1);
    }
    __syncthreads();                           // all waves done reading buf[cur]
  }
  int n_a = n0 + nb0 + l;
  int n_b = n_a + 16;
  float ba = qb[n_a], bb = qb[n_b];
  int mbase = m0 + mb + hi * 8;
#pragma unroll
  for (int r = 0; r < 8; ++r) {
    size_t rowoff = ((size_t)(b * NT_ + mbase + r)) * NO3;
    qkvT[rowoff + n_a] = (_Float16)(c0[r] + ba);
    qkvT[rowoff + n_b] = (_Float16)(c1[r] + bb);
  }
}

// ---------------- Kernel 4: flash attention ----------------
// block = (t-block of 128 queries, head, b); 8 waves, each owns 16 full query rows.
__global__ void attn_kernel(const _Float16* __restrict__ qkvT, _Float16* __restrict__ aT) {
  __shared__ __align__(16) _Float16 Qs[128][64];       // [t][c]
  __shared__ __align__(16) _Float16 Ks[64][64];        // [s][c]
  __shared__ __align__(16) _Float16 Vs[64][64];        // [c][s] (transposed)
  __shared__ __align__(16) _Float16 Ps[8][16][64];     // per-wave P tile [m][s]
  int t0 = blockIdx.x * 128, head = blockIdx.y, b = blockIdx.z;
  int tid = threadIdx.x, wave = tid >> 5, lane = tid & 31, l = lane & 15, hi = lane >> 4;
  int qo = head * 192, ko = qo + 64, vo = qo + 128;
  const _Float16* base = qkvT + (size_t)b * NT_ * NO3;

  // Q tile: 512 b128 chunks, async
#pragma unroll
  for (int idx = tid; idx < 512; idx += 256) {
    int r = idx >> 2, c8 = (idx & 3) * 16;             // 128 rows x 4 chunks
    g2l_b128(&Qs[r][c8], base + (size_t)(t0 + r) * NO3 + qo + c8);
  }
  WAIT_ASYNC(0);
  __syncthreads();

  const _Float16* qrow = &Qs[wave * 16 + l][0];
  v16h aq0 = ldA(qrow, 0, hi);
  v16h aq1 = ldA(qrow, 32, hi);

  v8f acc[4] = {};
  float mrow[8], lrow[8];
#pragma unroll
  for (int r = 0; r < 8; ++r) { mrow[r] = -3.0e38f; lrow[r] = 0.f; }

  for (int s0 = 0; s0 < NT_; s0 += 64) {
    __syncthreads();                                   // prev iter readers done
    // K tile: 256 b128 chunks -> 1 per thread, async
    {
      int r = tid >> 2, c8 = (tid & 3) * 16;
      g2l_b128(&Ks[r][c8], base + (size_t)(s0 + r) * NO3 + ko + c8);
    }
    // V tile with transpose (VGPR path)
#pragma unroll
    for (int idx = tid; idx < 64 * 16; idx += 256) {
      int r = idx >> 4, c4 = (idx & 15) * 4;
      v4h vv = *(const v4h*)(base + (size_t)(s0 + r) * NO3 + vo + c4);
      Vs[c4 + 0][r] = vv[0]; Vs[c4 + 1][r] = vv[1];
      Vs[c4 + 2][r] = vv[2]; Vs[c4 + 3][r] = vv[3];
    }
    WAIT_ASYNC(0);
    __syncthreads();

    // S = Q K^T  (wave: 16 rows x 64 keys)
    v8f sf[4] = {};
#pragma unroll
    for (int nb = 0; nb < 4; ++nb) {
      const _Float16* krow = &Ks[nb * 16 + l][0];
      sf[nb] = wmma_f16(aq0, ldB(krow, 0, hi), sf[nb]);
      sf[nb] = wmma_f16(aq1, ldB(krow, 32, hi), sf[nb]);
    }
#pragma unroll
    for (int nb = 0; nb < 4; ++nb)
#pragma unroll
      for (int r = 0; r < 8; ++r) sf[nb][r] *= 0.125f;   // 1/sqrt(ch)

    // online softmax; slot r = full query row spread over 16 lanes
#pragma unroll
    for (int r = 0; r < 8; ++r) {
      float mx = fmaxf(fmaxf(sf[0][r], sf[1][r]), fmaxf(sf[2][r], sf[3][r]));
      mx = fmaxf(mx, __shfl_xor(mx, 1, 32));
      mx = fmaxf(mx, __shfl_xor(mx, 2, 32));
      mx = fmaxf(mx, __shfl_xor(mx, 4, 32));
      mx = fmaxf(mx, __shfl_xor(mx, 8, 32));
      float mnew = fmaxf(mrow[r], mx);
      float corr = __expf(mrow[r] - mnew);
      mrow[r] = mnew;
      lrow[r] *= corr;
#pragma unroll
      for (int nb = 0; nb < 4; ++nb) acc[nb][r] *= corr;
      float rsum = 0.f;
#pragma unroll
      for (int nb = 0; nb < 4; ++nb) {
        float p = __expf(sf[nb][r] - mnew);
        sf[nb][r] = p;
        rsum += p;
      }
      rsum += __shfl_xor(rsum, 1, 32);
      rsum += __shfl_xor(rsum, 2, 32);
      rsum += __shfl_xor(rsum, 4, 32);
      rsum += __shfl_xor(rsum, 8, 32);
      lrow[r] += rsum;
      int m = hi * 8 + r;                                // D-layout -> LDS [m][s]
#pragma unroll
      for (int nb = 0; nb < 4; ++nb) Ps[wave][m][nb * 16 + l] = (_Float16)sf[nb][r];
    }
    __syncthreads();

    // acc += P V^T
    const _Float16* prow = &Ps[wave][l][0];
    v16h ap0 = ldA(prow, 0, hi);
    v16h ap1 = ldA(prow, 32, hi);
#pragma unroll
    for (int nb = 0; nb < 4; ++nb) {
      const _Float16* vrow = &Vs[nb * 16 + l][0];
      acc[nb] = wmma_f16(ap0, ldB(vrow, 0, hi), acc[nb]);
      acc[nb] = wmma_f16(ap1, ldB(vrow, 32, hi), acc[nb]);
    }
  }

  int mbase = t0 + wave * 16 + hi * 8;
#pragma unroll
  for (int r = 0; r < 8; ++r) {
    float inv = 1.0f / lrow[r];
    _Float16* dst = aT + ((size_t)(b * NT_ + mbase + r)) * NC_ + head * NCH;
#pragma unroll
    for (int nb = 0; nb < 4; ++nb) dst[nb * 16 + l] = (_Float16)(acc[nb][r] * inv);
  }
}

// ---------------- Kernel 5: proj GEMM + bias + residual ----------------
__global__ void proj_gemm(const _Float16* __restrict__ aT, const _Float16* __restrict__ pw,
                          const float* __restrict__ pb, const float* __restrict__ x,
                          float* __restrict__ out) {
  __shared__ __align__(16) _Float16 As[2][64][64];
  __shared__ __align__(16) _Float16 Bs[2][64][64];
  int n0 = blockIdx.x * 64, m0 = blockIdx.y * 64, b = blockIdx.z;
  int tid = threadIdx.x, wave = tid >> 5, lane = tid & 31, l = lane & 15, hi = lane >> 4;
  int mb = (wave >> 1) * 16, nb0 = (wave & 1) * 32;
  const _Float16* Ag = aT + ((size_t)(b * NT_ + m0)) * NC_;
  const _Float16* Bg = pw + (size_t)n0 * NC_;

  auto issue = [&](int buf, int k0) {
#pragma unroll
    for (int idx = tid; idx < 512; idx += 256) {
      int r = idx >> 3, c8 = (idx & 7) * 8;
      g2l_b128(&As[buf][r][c8], Ag + (size_t)r * NC_ + k0 + c8);
      g2l_b128(&Bs[buf][r][c8], Bg + (size_t)r * NC_ + k0 + c8);
    }
  };

  issue(0, 0);
  v8f c0 = {}, c1 = {};
  const int NK = NC_ / 64;
  for (int kt = 0; kt < NK; ++kt) {
    int cur = kt & 1;
    if (kt + 1 < NK) { issue(cur ^ 1, (kt + 1) * 64); WAIT_ASYNC(4); }
    else             { WAIT_ASYNC(0); }
    __syncthreads();
    const _Float16* arow  = &As[cur][mb + l][0];
    const _Float16* brow0 = &Bs[cur][nb0 + l][0];
    const _Float16* brow1 = &Bs[cur][nb0 + 16 + l][0];
#pragma unroll
    for (int kk = 0; kk < 64; kk += 32) {
      v16h af  = ldA(arow, kk, hi);
      v16h bf0 = ldB(brow0, kk, hi);
      v16h bf1 = ldB(brow1, kk, hi);
      c0 = wmma_f16(af, bf0, c0);
      c1 = wmma_f16(af, bf1, c1);
    }
    __syncthreads();
  }
  int n_a = n0 + nb0 + l;
  int n_b = n_a + 16;
  float ba = pb[n_a], bb = pb[n_b];
  int mbase = m0 + mb + hi * 8;
  size_t offa = ((size_t)(b * NC_ + n_a)) * NT_ + mbase;
  size_t offb = ((size_t)(b * NC_ + n_b)) * NT_ + mbase;
#pragma unroll
  for (int r = 0; r < 8; ++r) {
    out[offa + r] = x[offa + r] + c0[r] + ba;
    out[offb + r] = x[offb + r] + c1[r] + bb;
  }
}

// ---------------- launch ----------------
extern "C" void kernel_launch(void* const* d_in, const int* in_sizes, int n_in,
                              void* d_out, int out_size, void* d_ws, size_t ws_size,
                              hipStream_t stream) {
  const float* x      = (const float*)d_in[0];
  const float* norm_w = (const float*)d_in[1];
  const float* norm_b = (const float*)d_in[2];
  const float* qkv_w  = (const float*)d_in[3];
  const float* qkv_b  = (const float*)d_in[4];
  const float* proj_w = (const float*)d_in[5];
  const float* proj_b = (const float*)d_in[6];
  float* out = (float*)d_out;

  char* ws = (char*)d_ws;
  size_t off = 0;
  _Float16* xnT  = (_Float16*)(ws + off); off += (size_t)NB_ * NT_ * NC_ * 2;   // 8 MB
  _Float16* qwH  = (_Float16*)(ws + off); off += (size_t)NO3 * NC_ * 2;         // 1.5 MB
  _Float16* pwH  = (_Float16*)(ws + off); off += (size_t)NC_ * NC_ * 2;         // 0.5 MB
  _Float16* qkvT = (_Float16*)(ws + off); off += (size_t)NB_ * NT_ * NO3 * 2;   // 24 MB
  _Float16* aT   = (_Float16*)(ws + off); off += (size_t)NB_ * NT_ * NC_ * 2;   // 8 MB
  (void)ws_size; (void)n_in; (void)in_sizes; (void)out_size;

  gn_kernel<<<NB_ * 32, 256, 0, stream>>>(x, norm_w, norm_b, xnT);

  int nq = NO3 * NC_, np = NC_ * NC_;
  cvt_kernel<<<(nq + 255) / 256, 256, 0, stream>>>(qkv_w, qwH, nq);
  cvt_kernel<<<(np + 255) / 256, 256, 0, stream>>>(proj_w, pwH, np);

  qkv_gemm<<<dim3(NO3 / 64, NT_ / 64, NB_), 256, 0, stream>>>(xnT, qwH, qkv_b, qkvT);
  attn_kernel<<<dim3(NT_ / 128, NHEAD, NB_), 256, 0, stream>>>(qkvT, aT);
  proj_gemm<<<dim3(NC_ / 64, NT_ / 64, NB_), 256, 0, stream>>>(aT, pwH, proj_b, x, out);
}
